// MultiHeadSelfAttention_50586124812341
// MI455X (gfx1250) — matrix operand order repaired and checked by reading, not compiled, not run on hardware
//
#include <hip/hip_runtime.h>

// ---------------------------------------------------------------------------
// Performer (FAVOR+) self-attention forward, CDNA5 (gfx1250, wave32) WMMA.
// B=4, L=4096, E=768, H=12, D=64, M=128.  All GEMMs via v_wmma_f32_16x16x32_f16.
// ---------------------------------------------------------------------------

typedef __attribute__((ext_vector_type(16))) _Float16 v16h;
typedef __attribute__((ext_vector_type(8)))  _Float16 v8h;
typedef __attribute__((ext_vector_type(8)))  float    v8f;

#define R_ROWS   16384      // B*L
#define EMB      768
#define HEADS    12
#define DHEAD    64
#define MFEAT    128
#define SEQL     4096
#define BH       48         // B*H

__device__ __forceinline__ v8f wmma16(v16h a, v16h b, v8f c) {
  // D = A(16x32 f16) * B(32x16 f16) + C(16x16 f32)
  return __builtin_amdgcn_wmma_f32_16x16x32_f16(
      /*neg_a=*/false, a, /*neg_b=*/false, b,
      /*c_mod=*/(short)0, c, /*reuse_a=*/false, /*reuse_b=*/false);
}

// A fragment (16x32, row-major source, contiguous k order used by this file):
//   lanes 0-15 : row M=lane,  elems[0..15]=K0..15
//   lanes16-31 : row M=lane-16, elems[0..15]=K16..31
__device__ __forceinline__ v16h load_a_frag(const _Float16* __restrict__ A, int lda) {
  int lane = threadIdx.x & 31;
  int r    = lane & 15;
  int hf   = lane >> 4;
  const v8h* p0 = reinterpret_cast<const v8h*>(A + (size_t)r * lda + hf * 16);
  const v8h* p1 = reinterpret_cast<const v8h*>(A + (size_t)r * lda + 8 + hf * 16);
  v8h lo = *p0, hi = *p1;
  v16h f;
#pragma unroll
  for (int i = 0; i < 8; ++i) { f[i] = lo[i]; f[8 + i] = hi[i]; }
  return f;
}

// B fragment from pre-packed buffer: tile stored as 32 lanes x 16 halfs (coalesced 32B/lane).
__device__ __forceinline__ v16h load_b_packed(const _Float16* __restrict__ Bp) {
  int lane = threadIdx.x & 31;
  return reinterpret_cast<const v16h*>(Bp)[lane];
}

// ---------------------------------------------------------------------------
__global__ void f32_to_f16_kernel(const float* __restrict__ src,
                                  _Float16* __restrict__ dst, long n) {
  long i = (long)blockIdx.x * blockDim.x + threadIdx.x;
  long stride = (long)gridDim.x * blockDim.x;
  for (; i < n; i += stride) dst[i] = (_Float16)src[i];
}

// Pack f32 matrix (element (k,n) at src[k*sk + n*sn]) into WMMA-B fragment order.
// dst half index = ((nt*KT + kt)*32 + lane)*16 + i ;  k=kt*32+(lane>>4)*16+i ; n=nt*16+(lane&15)
__global__ void pack_b_kernel(const float* __restrict__ src, _Float16* __restrict__ dst,
                              int K, int N, int sk, int sn, float scale,
                              long srcBatchStride, long dstBatchStride) {
  src += (long)blockIdx.y * srcBatchStride;
  dst += (long)blockIdx.y * dstBatchStride;
  int total = K * N;
  int KT = K >> 5;
  for (int tid = blockIdx.x * blockDim.x + threadIdx.x; tid < total;
       tid += gridDim.x * blockDim.x) {
    int i    = tid & 15;
    int lane = (tid >> 4) & 31;
    int tile = tid >> 9;
    int kt = tile % KT, nt = tile / KT;
    int k = kt * 32 + (lane >> 4) * 16 + i;
    int n = nt * 16 + (lane & 15);
    dst[tid] = (_Float16)(src[(long)k * sk + (long)n * sn] * scale);
  }
}

// Pack per-(b,h) v slice [K=4096, N=64] (f16, row stride 768) into fragment order.
// grid = (1024, 48)
__global__ void pack_v_kernel(const _Float16* __restrict__ vh, _Float16* __restrict__ dst) {
  int bh = blockIdx.y;
  int b = bh / HEADS, h = bh % HEADS;
  const int total = DHEAD * SEQL;  // 262144 halfs per bh (KT=128, NT=4)
  dst += (size_t)bh * total;
  for (int tid = blockIdx.x * blockDim.x + threadIdx.x; tid < total;
       tid += gridDim.x * blockDim.x) {
    int i    = tid & 15;
    int lane = (tid >> 4) & 31;
    int tile = tid >> 9;
    int kt = tile & 127;   // KT = 128
    int nt = tile >> 7;
    int k = kt * 32 + (lane >> 4) * 16 + i;
    int n = nt * 16 + (lane & 15);
    dst[tid] = vh[(size_t)(b * SEQL + k) * EMB + h * DHEAD + n];
  }
}

// ---------------------------------------------------------------------------
// C = A(f16,[M x K] lda) * Bpacked(f16) + bias ; out f16 or f32, [M x ldo]
// block = 256 (8 waves); wave w: rows [bx*128 + w*16, +16), cols [by*64, +64)
template <bool OUT_F16>
__global__ __launch_bounds__(256) void gemm_bias_kernel(
    const _Float16* __restrict__ A, int lda,
    const _Float16* __restrict__ Bp,
    const float* __restrict__ bias,
    void* __restrict__ Out, int ldo, int K) {
  const int KT  = K >> 5;
  int wv   = threadIdx.x >> 5;
  int lane = threadIdx.x & 31;
  int row0 = blockIdx.x * 128 + wv * 16;
  int ntB  = blockIdx.y * 4;
  const _Float16* Arow = A + (size_t)row0 * lda;
  v8f acc[4] = {};
  for (int kt = 0; kt < KT; ++kt) {
    __builtin_prefetch(Arow + (size_t)(kt + 4) * 32, 0, 0);
    v16h a = load_a_frag(Arow + kt * 32, lda);
#pragma unroll
    for (int t = 0; t < 4; ++t) {
      v16h b = load_b_packed(Bp + ((size_t)(ntB + t) * KT + kt) * 512);
      acc[t] = wmma16(a, b, acc[t]);
    }
  }
  int n0 = lane & 15, mo = (lane >> 4) * 8;
#pragma unroll
  for (int t = 0; t < 4; ++t) {
    int n = (ntB + t) * 16 + n0;
    float bb = bias[n];
#pragma unroll
    for (int j = 0; j < 8; ++j) {
      int m = row0 + mo + j;
      float v = acc[t][j] + bb;
      if (OUT_F16) ((_Float16*)Out)[(size_t)m * ldo + n] = (_Float16)v;
      else         ((float*)Out)[(size_t)m * ldo + n]    = v;
    }
  }
}

// ---------------------------------------------------------------------------
// Performer feature map:  dd = (c*q) @ proj^T ; out = ratio*(exp(dd - diag - rowmax) + eps)
// Pp is packed projT already scaled by c = D^{-1/4}.  grid=(rows/128, H)
// mode 0: write q' to outQ [R, H*M] row-major
// mode 1: write k' transposed to outKT [B*H][M][L]
__global__ __launch_bounds__(256) void feat_kernel(
    const _Float16* __restrict__ QK,   // [R, 768]
    const _Float16* __restrict__ Pp,   // packed: NT=8, KT=2
    _Float16* __restrict__ outQ,
    _Float16* __restrict__ outKT,
    int mode) {
  const float RATIO = 0.08838834764831845f;  // 1/sqrt(128)
  const float EPSC  = 1e-6f;
  int h    = blockIdx.y;
  int wv   = threadIdx.x >> 5;
  int lane = threadIdx.x & 31;
  int row0 = blockIdx.x * 128 + wv * 16;
  const _Float16* Arow = QK + (size_t)row0 * EMB + h * DHEAD;

  v8f acc[8] = {};
#pragma unroll
  for (int kt = 0; kt < 2; ++kt) {
    v16h a = load_a_frag(Arow + kt * 32, EMB);
#pragma unroll
    for (int t = 0; t < 8; ++t) {
      v16h b = load_b_packed(Pp + ((size_t)t * 2 + kt) * 512);
      acc[t] = wmma16(a, b, acc[t]);
    }
  }

  // diag = (c^2/2)*sum(q^2) = sumsq/16 per row, via LDS (16 rows per wave)
  __shared__ float diagLds[8][16];
  if (lane < 16) {
    const v8h* r = reinterpret_cast<const v8h*>(QK + (size_t)(row0 + lane) * EMB + h * DHEAD);
    float s = 0.f;
#pragma unroll
    for (int c = 0; c < 8; ++c) {
      v8h q = r[c];
#pragma unroll
      for (int i = 0; i < 8; ++i) { float f = (float)q[i]; s += f * f; }
    }
    diagLds[wv][lane] = s * (1.0f / 16.0f);
  }
  __syncthreads();

  // row-wise max across the 128 features (stab). Row (grp*8+j) lives in vgpr j of lanes of grp.
  float stab[8];
#pragma unroll
  for (int j = 0; j < 8; ++j) {
    float mx = acc[0][j];
#pragma unroll
    for (int t = 1; t < 8; ++t) mx = fmaxf(mx, acc[t][j]);
    mx = fmaxf(mx, __shfl_xor(mx, 1));
    mx = fmaxf(mx, __shfl_xor(mx, 2));
    mx = fmaxf(mx, __shfl_xor(mx, 4));
    mx = fmaxf(mx, __shfl_xor(mx, 8));
    stab[j] = mx;
  }

  int n0 = lane & 15, mo = (lane >> 4) * 8;
#pragma unroll
  for (int j = 0; j < 8; ++j) {
    float dg = diagLds[wv][mo + j];
    int rg   = row0 + mo + j;                  // global (b,l) row
#pragma unroll
    for (int t = 0; t < 8; ++t) {
      float v = RATIO * (__expf(acc[t][j] - dg - stab[j]) + EPSC);
      int feat = t * 16 + n0;
      if (mode == 0) {
        outQ[(size_t)rg * (HEADS * MFEAT) + h * MFEAT + feat] = (_Float16)v;
      } else {
        int b = rg >> 12, l = rg & (SEQL - 1);
        outKT[((size_t)(b * HEADS + h) * MFEAT + feat) * SEQL + l] = (_Float16)v;
      }
    }
  }
}

// ---------------------------------------------------------------------------
// kvs_aug[bh][m][0..63] = sum_l k'[bh][m][l] * v[b][l][h*64+d]
// kvs_aug[bh][m][64]    = sum_l k'[bh][m][l]      (ones column)
// grid = 48 blocks, 8 waves; wave = one 16-row M-tile, all 5 N-tiles, K=4096.
__global__ __launch_bounds__(256) void kvs_kernel(
    const _Float16* __restrict__ kpT,   // [48][128][4096]
    const _Float16* __restrict__ pVv,   // packed per bh: NT=4, KT=128 (4*128*512 halfs)
    float* __restrict__ kvs) {          // [48][128][80]
  int bh = blockIdx.x;
  int wv = threadIdx.x >> 5, lane = threadIdx.x & 31;
  int m0 = wv * 16;
  const _Float16* Arow = kpT + ((size_t)bh * MFEAT + m0) * SEQL;
  const _Float16* Bp   = pVv + (size_t)bh * (DHEAD * SEQL);

  v16h ones;
#pragma unroll
  for (int i = 0; i < 16; ++i) ones[i] = (_Float16)0.0f;
  if ((lane & 15) == 0) {
#pragma unroll
    for (int i = 0; i < 16; ++i) ones[i] = (_Float16)1.0f;
  }

  v8f acc[5] = {};
  for (int kt = 0; kt < SEQL / 32; ++kt) {
    __builtin_prefetch(Arow + (size_t)(kt + 4) * 32, 0, 0);
    v16h a = load_a_frag(Arow + kt * 32, SEQL);
#pragma unroll
    for (int t = 0; t < 4; ++t) {
      v16h bfr = load_b_packed(Bp + ((size_t)(t * 128 + kt)) * 512);
      acc[t] = wmma16(a, bfr, acc[t]);
    }
    acc[4] = wmma16(a, ones, acc[4]);
  }
  int n0 = lane & 15, mo = (lane >> 4) * 8;
#pragma unroll
  for (int t = 0; t < 5; ++t)
#pragma unroll
    for (int j = 0; j < 8; ++j)
      kvs[((size_t)bh * MFEAT + m0 + mo + j) * 80 + t * 16 + n0] = acc[t][j];
}

// ---------------------------------------------------------------------------
// num/den: A = q'[b, l0.., h*128..] (K=128), B = packed kvs_aug (N=80, col64=ks_sum)
// attention = num/den written straight into concat layout [B,H,L,D] flattened.
__global__ __launch_bounds__(256) void numden_kernel(
    const _Float16* __restrict__ Qp,    // [16384][1536]
    const _Float16* __restrict__ Kvp,   // packed per bh: 5*4*512 halfs
    _Float16* __restrict__ concat) {    // [16384][768] in [B,H,L,D] order
  int bh = blockIdx.y;
  int b = bh / HEADS, h = bh % HEADS;
  int wv = threadIdx.x >> 5, lane = threadIdx.x & 31;
  int l0 = blockIdx.x * 128 + wv * 16;
  const _Float16* Arow = Qp + ((size_t)(b * SEQL + l0)) * (HEADS * MFEAT) + h * MFEAT;
  const _Float16* Bp = Kvp + (size_t)bh * (5 * 4 * 512);

  v8f acc[5] = {};
#pragma unroll
  for (int kt = 0; kt < 4; ++kt) {
    v16h a = load_a_frag(Arow + kt * 32, HEADS * MFEAT);
#pragma unroll
    for (int t = 0; t < 5; ++t) {
      v16h bfr = load_b_packed(Bp + ((size_t)t * 4 + kt) * 512);
      acc[t] = wmma16(a, bfr, acc[t]);
    }
  }

  int n0 = lane & 15, mo = (lane >> 4) * 8;
  int srcLane = (lane < 16) ? 0 : 16;  // n_local==0 lanes hold column 64 (den)
#pragma unroll
  for (int j = 0; j < 8; ++j) {
    float den = __shfl(acc[4][j], srcLane, 32);
    float inv = 1.0f / den;
    int l = l0 + mo + j;
    size_t base = ((size_t)bh * SEQL + l) * DHEAD;
#pragma unroll
    for (int t = 0; t < 4; ++t)
      concat[base + t * 16 + n0] = (_Float16)(acc[t][j] * inv);
  }
}

// ---------------------------------------------------------------------------
extern "C" void kernel_launch(void* const* d_in, const int* in_sizes, int n_in,
                              void* d_out, int out_size, void* d_ws, size_t ws_size,
                              hipStream_t stream) {
  (void)in_sizes; (void)n_in; (void)out_size; (void)ws_size;
  const float* x  = (const float*)d_in[0];
  const float* Wq = (const float*)d_in[1];
  const float* bq = (const float*)d_in[2];
  const float* Wk = (const float*)d_in[3];
  const float* bk = (const float*)d_in[4];
  const float* Wv = (const float*)d_in[5];
  const float* bv = (const float*)d_in[6];
  const float* Wo = (const float*)d_in[7];
  const float* bo = (const float*)d_in[8];
  const float* pj = (const float*)d_in[9];

  char* ws = (char*)d_ws;
  size_t off = 0;
  auto take = [&](size_t bytes) -> char* {
    char* p = ws + off;
    off += (bytes + 255) & ~(size_t)255;
    return p;
  };

  _Float16* xh  = (_Float16*)take((size_t)R_ROWS * EMB * 2);      // x f16, reused as concat
  _Float16* qh  = (_Float16*)take((size_t)R_ROWS * EMB * 2);      // reused as packed-v
  _Float16* kh  = (_Float16*)take((size_t)R_ROWS * EMB * 2);
  _Float16* vh  = (_Float16*)take((size_t)R_ROWS * EMB * 2);
  _Float16* qp  = (_Float16*)take((size_t)R_ROWS * HEADS * MFEAT * 2);
  _Float16* kpT = (_Float16*)take((size_t)BH * MFEAT * SEQL * 2);
  _Float16* pWq = (_Float16*)take((size_t)EMB * EMB * 2);
  _Float16* pWk = (_Float16*)take((size_t)EMB * EMB * 2);
  _Float16* pWv = (_Float16*)take((size_t)EMB * EMB * 2);
  _Float16* pWo = (_Float16*)take((size_t)EMB * EMB * 2);
  _Float16* pPj = (_Float16*)take((size_t)DHEAD * MFEAT * 2);
  float*    kvs = (float*)take((size_t)BH * MFEAT * 80 * 4);
  _Float16* pKv = (_Float16*)take((size_t)BH * MFEAT * 80 * 2);
  _Float16* concat = xh;  // x no longer needed after QKV GEMMs
  _Float16* pVv    = qh;  // qh no longer needed after feat(q); 48*64*4096 == 16384*768

  const float CN = 0.35355339059327373f;  // 64^{-1/4}, folded into projT

  // 1) convert x -> f16
  f32_to_f16_kernel<<<4096, 256, 0, stream>>>(x, xh, (long)R_ROWS * EMB);

  // 2) pack weights + scaled projT into WMMA-B fragment order
  pack_b_kernel<<<dim3(2304, 1), 256, 0, stream>>>(Wq, pWq, EMB, EMB, EMB, 1, 1.0f, 0, 0);
  pack_b_kernel<<<dim3(2304, 1), 256, 0, stream>>>(Wk, pWk, EMB, EMB, EMB, 1, 1.0f, 0, 0);
  pack_b_kernel<<<dim3(2304, 1), 256, 0, stream>>>(Wv, pWv, EMB, EMB, EMB, 1, 1.0f, 0, 0);
  pack_b_kernel<<<dim3(2304, 1), 256, 0, stream>>>(Wo, pWo, EMB, EMB, EMB, 1, 1.0f, 0, 0);
  // projT[d,m] = proj[m,d] -> sk=1, sn=64, scaled by CN
  pack_b_kernel<<<dim3(32, 1), 256, 0, stream>>>(pj, pPj, DHEAD, MFEAT, 1, DHEAD, CN, 0, 0);

  // 3) q,k,v projections (f16 out, f32 accumulate, +bias)
  gemm_bias_kernel<true><<<dim3(128, 12), 256, 0, stream>>>(xh, EMB, pWq, bq, qh, EMB, EMB);
  gemm_bias_kernel<true><<<dim3(128, 12), 256, 0, stream>>>(xh, EMB, pWk, bk, kh, EMB, EMB);
  gemm_bias_kernel<true><<<dim3(128, 12), 256, 0, stream>>>(xh, EMB, pWv, bv, vh, EMB, EMB);

  // 4) Performer features: q' first (frees qh), then k' (transposed [bh][m][l])
  feat_kernel<<<dim3(128, 12), 256, 0, stream>>>(qh, pPj, qp, kpT, 0);
  feat_kernel<<<dim3(128, 12), 256, 0, stream>>>(kh, pPj, qp, kpT, 1);

  // 5) pack per-head v into fragment order (reuses qh region)
  pack_v_kernel<<<dim3(1024, BH), 256, 0, stream>>>(vh, pVv);

  // 6) kvs (+ ks_sum as column 64)
  kvs_kernel<<<dim3(BH), 256, 0, stream>>>(kpT, pVv, kvs);

  // 7) pack kvs_aug fragments (batched over bh)
  pack_b_kernel<<<dim3(40, BH), 256, 0, stream>>>(kvs, pKv, MFEAT, 80, 80, 1, 1.0f,
                                                  (long)MFEAT * 80, (long)MFEAT * 80);

  // 8) num/den + divide + transposed concat
  numden_kernel<<<dim3(32, BH), 256, 0, stream>>>(qp, pKv, concat);

  // 9) out = concat @ Wo + bo (f32)
  gemm_bias_kernel<false><<<dim3(128, 12), 256, 0, stream>>>(concat, EMB, pWo, bo,
                                                             d_out, EMB, EMB);
}
